// GramSpecRouter_44693429682385
// MI455X (gfx1250) — matrix operand order repaired and checked by reading, not compiled, not run on hardware
//
#include <hip/hip_runtime.h>

// ---------------------------------------------------------------------------
// GramSpecRouter for MI455X (gfx1250, wave32, WMMA, async-to-LDS staging).
//
//   K0 init_convert : w_hh, expr_w -> bf16 in ws; zero scalar accumulators
//   K1 gemm_bf16    : XI  = x @ w_ih^T   [8192 x 3072]  WMMA + async LDS staging
//   K2 gemm_bf16    : EXPR= x @ expr_w^T [8192 x 1024]  -> d_out
//   K3 gru_scan     : 2048 sequential steps, h double-buffered in LDS, WMMA
//   K4 row_norms    : ||expr_w row||
//   K5 expr_loss    : mean((l2norm(W) l2norm(W)^T - I)^2) via WMMA + atomicAdd
//   K6 gram_cos     : per (b,s) Gram / speciality partials / cosine sims
//   K7 finalize_pen : penalty = acc / (B*S)
//   K8 topk_softmax : top-2 + softmax -> multiplier / selected experts
//
// Workspace: XI 100.66MB | routing 33.55MB | w_hh bf16 6.29MB |
//            expr_w bf16 4.19MB | norms 4KB | pen acc  => ~144.8MB
// ---------------------------------------------------------------------------

#define Bb   4
#define Ss   2048
#define Hh   2048
#define Ee   8
#define Dd   128
#define GH   1024
#define N3   3072
#define BS   8192

// d_out offsets (floats)
#define OFF_MULT   0u
#define OFF_SEL    16384u
#define OFF_EXPR   32768u
#define OFF_HLAST  8421376u
#define OFF_PEN    8425472u
#define OFF_COS    8425473u
#define OFF_ELOSS  8491009u

// ws offsets (bytes)
#define WS_XI     0ull
#define WS_ROUT   100663296ull
#define WS_WHH    134217728ull
#define WS_WEXP   140509184ull
#define WS_NORM   144703488ull
#define WS_PEN    144707584ull

typedef __attribute__((ext_vector_type(16))) __bf16          v16bf;
typedef __attribute__((ext_vector_type(16))) unsigned short  v16u16;
typedef __attribute__((ext_vector_type(8)))  float           v8f;

// ---- gfx1250 async global->LDS path (guarded; sync fallback) ---------------
#if __has_builtin(__builtin_amdgcn_global_load_async_to_lds_b128) && \
    __has_builtin(__builtin_amdgcn_s_wait_asynccnt)
#define GSR_ASYNC 1
#else
#define GSR_ASYNC 0
#endif

#if GSR_ASYNC
// builtin expects (int4 AS1*, int4 AS3*, imm offset, imm cpol)
typedef int gsr_i4 __attribute__((vector_size(16)));
typedef __attribute__((address_space(1))) gsr_i4 gsr_i4_g;
typedef __attribute__((address_space(3))) gsr_i4 gsr_i4_l;
#define WAIT_ASYNC(n) __builtin_amdgcn_s_wait_asynccnt(n)
#else
#define WAIT_ASYNC(n) ((void)0)
#endif

// 16-byte global->LDS copy (async DMA when available)
__device__ __forceinline__ void async_cp16(const float* gp, float* lp) {
#if GSR_ASYNC
  gsr_i4_g* g = (gsr_i4_g*)(unsigned long long)(size_t)gp;
  gsr_i4_l* l = (gsr_i4_l*)(unsigned int)(size_t)lp;  // low 32b of shared addr = LDS offset
  __builtin_amdgcn_global_load_async_to_lds_b128(g, l, 0, 0);
#else
  *(float4*)lp = *(const float4*)gp;
#endif
}

// ---- fragment builders (ISA 7.12.2 layouts, wave32) ------------------------
__device__ __forceinline__ void cvt4bf(v16bf& t, int base, float4 q) {
  t[base + 0] = (__bf16)q.x;
  t[base + 1] = (__bf16)q.y;
  t[base + 2] = (__bf16)q.z;
  t[base + 3] = (__bf16)q.w;
}

union U16x16 { uint4 q[2]; v16u16 v; };

// A (16-bit, 16x32, MxK): lane L: row=L&15, half=L>>4.
//   elems 0..7  <- K = half*8 + (0..7);  elems 8..15 <- K = 16 + half*8 + (0..7)
__device__ __forceinline__ v16bf frag_a_f32(const float* __restrict__ base,
                                            int ld, int r0, int k0, int lane) {
  int half = lane >> 4, r = lane & 15;
  const float* p = base + (size_t)(r0 + r) * (size_t)ld + (size_t)(k0 + half * 8);
  v16bf t;
  cvt4bf(t, 0,  *reinterpret_cast<const float4*>(p));
  cvt4bf(t, 4,  *reinterpret_cast<const float4*>(p + 4));
  cvt4bf(t, 8,  *reinterpret_cast<const float4*>(p + 16));
  cvt4bf(t, 12, *reinterpret_cast<const float4*>(p + 20));
  return t;
}

__device__ __forceinline__ v16bf frag_a_u16(const unsigned short* __restrict__ base,
                                            int ld, int r0, int k0, int lane) {
  int half = lane >> 4, r = lane & 15;
  const unsigned short* p = base + (size_t)(r0 + r) * (size_t)ld + (size_t)(k0 + half * 8);
  U16x16 u;
  u.q[0] = *reinterpret_cast<const uint4*>(p);
  u.q[1] = *reinterpret_cast<const uint4*>(p + 16);
  return __builtin_bit_cast(v16bf, u.v);
}

// B (16-bit, 32x16, KxN): lane L: col=L&15, half=L>>4; elems 0..7 <- K = half*8 + (0..7); elems 8..15 <- K = 16 + half*8 + (0..7).
// B[k,n] = W[n,k] for W row-major [N,K].
__device__ __forceinline__ v16bf frag_b_f32(const float* __restrict__ W,
                                            int ld, int n0, int k0, int lane) {
  int half = lane >> 4, n = lane & 15;
  const float* p = W + (size_t)(n0 + n) * (size_t)ld + (size_t)(k0 + half * 8);
  v16bf t;
  cvt4bf(t, 0,  *reinterpret_cast<const float4*>(p));
  cvt4bf(t, 4,  *reinterpret_cast<const float4*>(p + 4));
  cvt4bf(t, 8,  *reinterpret_cast<const float4*>(p + 16));
  cvt4bf(t, 12, *reinterpret_cast<const float4*>(p + 20));
  return t;
}

__device__ __forceinline__ v16bf frag_b_u16(const unsigned short* __restrict__ W,
                                            int ld, int n0, int k0, int lane) {
  int half = lane >> 4, n = lane & 15;
  const unsigned short* p = W + (size_t)(n0 + n) * (size_t)ld + (size_t)(k0 + half * 8);
  U16x16 u;
  u.q[0] = *reinterpret_cast<const uint4*>(p);
  u.q[1] = *reinterpret_cast<const uint4*>(p + 16);
  return __builtin_bit_cast(v16bf, u.v);
}

#define WMMA_BF16(a, b, c) \
  __builtin_amdgcn_wmma_f32_16x16x32_bf16(false, (a), false, (b), (short)0, (c), false, false)

// ---------------------------------------------------------------------------
// K0: convert weights to bf16, zero accumulators
// ---------------------------------------------------------------------------
__global__ void k_init_convert(const float* __restrict__ whh,
                               const float* __restrict__ wexp,
                               unsigned short* __restrict__ whh_bf,
                               unsigned short* __restrict__ wexp_bf,
                               float* __restrict__ pen_acc,
                               float* __restrict__ out_eloss) {
  size_t i = (size_t)blockIdx.x * blockDim.x + threadIdx.x;
  size_t stride = (size_t)gridDim.x * blockDim.x;
  for (size_t j = i; j < (size_t)N3 * GH; j += stride)
    whh_bf[j] = __builtin_bit_cast(unsigned short, (__bf16)whh[j]);
  for (size_t j = i; j < (size_t)GH * Hh; j += stride)
    wexp_bf[j] = __builtin_bit_cast(unsigned short, (__bf16)wexp[j]);
  if (i == 0) { *pen_acc = 0.f; *out_eloss = 0.f; }
}

// ---------------------------------------------------------------------------
// K1/K2: C[8192, N] = A[8192, K] * Bw[N, K]^T.
// Double-buffered async HBM->LDS staging of 64x32 A / 128x32 B f32 panels,
// WMMA bf16 compute. Block = 8 waves (2x4), wave tile 32x32 (2x2 WMMA).
// ---------------------------------------------------------------------------
__global__ __launch_bounds__(256)
void k_gemm_bf16(const float* __restrict__ A, const float* __restrict__ Bw,
                 float* __restrict__ C, int N, int K) {
  __shared__ alignas(16) float sA[2][64 * 32];
  __shared__ alignas(16) float sB[2][128 * 32];

  int tid = threadIdx.x;
  int lane = tid & 31, wave = tid >> 5;
  int wr = wave >> 2, wc = wave & 3;
  int rowB = blockIdx.x * 64;
  int colB = blockIdx.y * 128;

  // 6 x 16B async copies per thread per panel (A:2, B:4); ASYNCcnt completes
  // in order, so waiting <=6 drains the previous panel while the next flies.
  auto issue = [&](int buf, int k0) {
#pragma unroll
    for (int j = 0; j < 2; ++j) {
      int f = tid + j * 256;
      int row = f >> 3, c4 = (f & 7) * 4;
      async_cp16(A + (size_t)(rowB + row) * K + k0 + c4, &sA[buf][row * 32 + c4]);
    }
#pragma unroll
    for (int j = 0; j < 4; ++j) {
      int f = tid + j * 256;
      int row = f >> 3, c4 = (f & 7) * 4;
      async_cp16(Bw + (size_t)(colB + row) * K + k0 + c4, &sB[buf][row * 32 + c4]);
    }
  };

  v8f z = {0.f, 0.f, 0.f, 0.f, 0.f, 0.f, 0.f, 0.f};
  v8f acc[2][2];
  acc[0][0] = z; acc[0][1] = z; acc[1][0] = z; acc[1][1] = z;

  int nk = K / 32;
  issue(0, 0);

#pragma unroll 1
  for (int ks = 0; ks < nk; ++ks) {
    int cur = ks & 1;
    if (ks + 1 < nk) { issue(cur ^ 1, (ks + 1) * 32); WAIT_ASYNC(6); }
    else             { WAIT_ASYNC(0); }
    __syncthreads();

    int r0 = wr * 32, c0 = wc * 32;
    v16bf a0 = frag_a_f32(&sA[cur][0], 32, r0,      0, lane);
    v16bf a1 = frag_a_f32(&sA[cur][0], 32, r0 + 16, 0, lane);
    v16bf b0 = frag_b_f32(&sB[cur][0], 32, c0,      0, lane);
    v16bf b1 = frag_b_f32(&sB[cur][0], 32, c0 + 16, 0, lane);
    acc[0][0] = WMMA_BF16(a0, b0, acc[0][0]);
    acc[0][1] = WMMA_BF16(a0, b1, acc[0][1]);
    acc[1][0] = WMMA_BF16(a1, b0, acc[1][0]);
    acc[1][1] = WMMA_BF16(a1, b1, acc[1][1]);
    __syncthreads();   // protect buffer reuse at ks+1's issue()
  }

  int half = lane >> 4, cn = lane & 15;
#pragma unroll
  for (int tm = 0; tm < 2; ++tm)
#pragma unroll
    for (int tn = 0; tn < 2; ++tn) {
      float* cp = C + (size_t)(rowB + wr * 32 + tm * 16 + 8 * half) * (size_t)N
                    + (size_t)(colB + wc * 32 + tn * 16 + cn);
#pragma unroll
      for (int v = 0; v < 8; ++v) cp[(size_t)v * N] = acc[tm][tn][v];
    }
}

// ---------------------------------------------------------------------------
// K3: GRU scan. One workgroup, 32 waves, h[4,1024] double-buffered in LDS.
// Each wave owns 2 g-tiles (16 hidden units) x 3 gates; M padded 4->16.
// ---------------------------------------------------------------------------
__global__ __launch_bounds__(1024)
void k_gru_scan(const float* __restrict__ xi,             // [B*S, 3GH]
                const unsigned short* __restrict__ whh,   // [3GH, GH] bf16
                const float* __restrict__ h0,             // [B, GH]
                float* __restrict__ routing,              // [B*S, GH]
                float* __restrict__ h_last) {             // [B*GH]
  __shared__ alignas(16) float hbuf[2][4 * GH];
  int tid = threadIdx.x;
  int wave = tid >> 5, lane = tid & 31;
  int half = lane >> 4, cn = lane & 15;
  int gt0 = wave * 2;

  for (int i = tid; i < 4 * GH; i += 1024) hbuf[0][i] = h0[i];
  __syncthreads();

  v8f z = {0.f, 0.f, 0.f, 0.f, 0.f, 0.f, 0.f, 0.f};

#pragma unroll 1
  for (int t = 0; t < Ss; ++t) {
    int cur = t & 1, nxt = cur ^ 1;
    v8f acc[2][3];
#pragma unroll
    for (int gi = 0; gi < 2; ++gi)
#pragma unroll
      for (int q = 0; q < 3; ++q) acc[gi][q] = z;

#pragma unroll 1
    for (int k0 = 0; k0 < GH; k0 += 32) {
      v16bf afrag;
#pragma unroll
      for (int i = 0; i < 16; ++i) afrag[i] = (__bf16)0.f;
      int r = lane & 15;
      if (r < 4) {
        const float* hp = &hbuf[cur][r * GH + k0 + half * 8];
        cvt4bf(afrag, 0,  *reinterpret_cast<const float4*>(hp));
        cvt4bf(afrag, 4,  *reinterpret_cast<const float4*>(hp + 4));
        cvt4bf(afrag, 8,  *reinterpret_cast<const float4*>(hp + 16));
        cvt4bf(afrag, 12, *reinterpret_cast<const float4*>(hp + 20));
      }
#pragma unroll
      for (int gi = 0; gi < 2; ++gi)
#pragma unroll
        for (int q = 0; q < 3; ++q) {
          v16bf bfrag = frag_b_u16(whh, GH, q * GH + (gt0 + gi) * 16, k0, lane);
          acc[gi][q] = WMMA_BF16(afrag, bfrag, acc[gi][q]);
        }
    }

    if (half == 0) {
#pragma unroll
      for (int gi = 0; gi < 2; ++gi) {
        int g = (gt0 + gi) * 16 + cn;
#pragma unroll
        for (int b = 0; b < 4; ++b) {
          const float* xrow = xi + ((size_t)b * Ss + t) * (size_t)N3;
          float xr = xrow[g], xz = xrow[GH + g], xn = xrow[2 * GH + g];
          float hr = acc[gi][0][b], hz = acc[gi][1][b], hn_ = acc[gi][2][b];
          float rg = 1.f / (1.f + __expf(-(xr + hr)));
          float zg = 1.f / (1.f + __expf(-(xz + hz)));
          float ng = tanhf(xn + rg * hn_);
          float hp = hbuf[cur][b * GH + g];
          float hnew = (1.f - zg) * ng + zg * hp;
          hbuf[nxt][b * GH + g] = hnew;
          routing[((size_t)b * Ss + t) * (size_t)GH + g] = hnew;
        }
      }
    }
    __syncthreads();
  }

  for (int i = tid; i < 4 * GH; i += 1024) h_last[i] = hbuf[Ss & 1][i];
}

// ---------------------------------------------------------------------------
// K4: expr_w row norms (one wave per row)
// ---------------------------------------------------------------------------
__global__ __launch_bounds__(256)
void k_row_norms(const float* __restrict__ W, float* __restrict__ norms) {
  int gwave = (blockIdx.x * 256 + threadIdx.x) >> 5;
  int lane = threadIdx.x & 31;
  if (gwave >= GH) return;
  const float* r = W + (size_t)gwave * Hh;
  float s = 0.f;
  for (int i = lane; i < Hh; i += 32) { float v = r[i]; s += v * v; }
#pragma unroll
  for (int m = 16; m; m >>= 1) s += __shfl_xor(s, m, 32);
  if (lane == 0) norms[gwave] = fmaxf(sqrtf(s), 1e-12f);
}

// ---------------------------------------------------------------------------
// K5: expression loss via WMMA tiles + atomicAdd
// ---------------------------------------------------------------------------
__global__ __launch_bounds__(256)
void k_expr_loss(const unsigned short* __restrict__ Wbf,  // [GH, H] bf16
                 const float* __restrict__ norms,
                 float* __restrict__ out_loss) {
  int lane = threadIdx.x & 31, wave = threadIdx.x >> 5;
  int tile = blockIdx.x * 8 + wave;          // 4096 tiles (64x64)
  int i0 = (tile >> 6) * 16, j0 = (tile & 63) * 16;

  v8f c = {0.f, 0.f, 0.f, 0.f, 0.f, 0.f, 0.f, 0.f};
#pragma unroll 1
  for (int k0 = 0; k0 < Hh; k0 += 32) {
    v16bf a = frag_a_u16(Wbf, Hh, i0, k0, lane);
    v16bf b = frag_b_u16(Wbf, Hh, j0, k0, lane);
    c = WMMA_BF16(a, b, c);
  }

  int half = lane >> 4, cn = lane & 15;
  float local = 0.f;
#pragma unroll
  for (int v = 0; v < 8; ++v) {
    int i = i0 + v + 8 * half, j = j0 + cn;
    float g = c[v] / (norms[i] * norms[j]);
    float d = g - (i == j ? 1.f : 0.f);
    local += d * d;
  }
#pragma unroll
  for (int m = 16; m; m >>= 1) local += __shfl_xor(local, m, 32);
  if (lane == 0) atomicAdd(out_loss, local * (1.f / (1024.f * 1024.f)));
}

// ---------------------------------------------------------------------------
// K6: per-(b,s) Gram / speciality partials / cosine similarities
// ---------------------------------------------------------------------------
__global__ __launch_bounds__(256)
void k_gram_cos(const float* __restrict__ routing,        // [BS, GH]
                const float* __restrict__ expr,           // [BS, GH]
                float* __restrict__ cos_out,              // [BS, E]
                float* __restrict__ pen_acc) {
  __shared__ alignas(16) float shR[8][GH];
  __shared__ float shG[8][64];
  __shared__ float shN[8][8];
  int wave = threadIdx.x >> 5, lane = threadIdx.x & 31;
  int idx = blockIdx.x * 8 + wave;

  const float* rrow = routing + (size_t)idx * GH;
  for (int j = lane; j < GH; j += 32) shR[wave][j] = rrow[j];

#pragma unroll
  for (int e = 0; e < Ee; ++e) {
    float s = 0.f;
    for (int d = lane; d < Dd; d += 32) { float v = shR[wave][e * Dd + d]; s += v * v; }
#pragma unroll
    for (int m = 16; m; m >>= 1) s += __shfl_xor(s, m, 32);
    if (lane == 0) shN[wave][e] = fmaxf(sqrtf(s), 1e-12f);
  }
  __syncthreads();

  for (int j = lane; j < GH; j += 32) shR[wave][j] /= shN[wave][j >> 7];
  __syncthreads();

  for (int p = lane; p < 64; p += 32) {
    int e = p >> 3, f = p & 7;
    float s = 0.f;
    for (int d = 0; d < Dd; ++d) s += shR[wave][e * Dd + d] * shR[wave][f * Dd + d];
    shG[wave][p] = s;
  }
  __syncthreads();

  float pen = 0.f;
  if (lane < Ee) {
    float s = 0.f;
#pragma unroll
    for (int f = 0; f < Ee; ++f) {
      float d = shG[wave][lane * 8 + f] - (lane == f ? 1.f : 0.f);
      s += d * d;
    }
    float rn = sqrtf(s);
    float m = fmaxf(rn, 1e-12f);
    pen = (rn * rn) / (m * m);
  }
#pragma unroll
  for (int m = 16; m; m >>= 1) pen += __shfl_xor(pen, m, 32);
  if (lane == 0) atomicAdd(pen_acc, pen);

  int e = lane >> 2, q = lane & 3;
  const float* erow = expr + (size_t)idx * GH + e * Dd;
  float dot = 0.f, na = 0.f, nb = 0.f;
  for (int i = 0; i < 32; ++i) {
    int d = q * 32 + i;
    float ev = erow[d], rv = shR[wave][e * Dd + d];
    dot += ev * rv; na += ev * ev; nb += rv * rv;
  }
  dot += __shfl_xor(dot, 1, 32); dot += __shfl_xor(dot, 2, 32);
  na  += __shfl_xor(na, 1, 32);  na  += __shfl_xor(na, 2, 32);
  nb  += __shfl_xor(nb, 1, 32);  nb  += __shfl_xor(nb, 2, 32);
  if (q == 0) {
    float c = dot / (fmaxf(sqrtf(na), 1e-8f) * fmaxf(sqrtf(nb), 1e-8f));
    cos_out[(size_t)idx * Ee + e] = 1.f - c;
  }
}

__global__ void k_finalize_pen(const float* __restrict__ pen_acc,
                               float* __restrict__ out_pen) {
  *out_pen = *pen_acc / (float)BS;
}

// ---------------------------------------------------------------------------
// K8: domain scores, top-2 (earliest-index ties like jax), softmax
// ---------------------------------------------------------------------------
__global__ __launch_bounds__(256)
void k_topk_softmax(const float* __restrict__ cosv, const float* __restrict__ pen,
                    float* __restrict__ mult, float* __restrict__ sel) {
  int idx = blockIdx.x * 256 + threadIdx.x;
  if (idx >= BS) return;
  float scale = 1.f + *pen;
  float s[Ee];
#pragma unroll
  for (int e = 0; e < Ee; ++e) s[e] = cosv[(size_t)idx * Ee + e] * scale;
  int i0 = 0;
#pragma unroll
  for (int e = 1; e < Ee; ++e) if (s[e] > s[i0]) i0 = e;
  int i1 = -1;
#pragma unroll
  for (int e = 0; e < Ee; ++e)
    if (e != i0 && (i1 < 0 || s[e] > s[i1])) i1 = e;
  float v0 = s[i0], v1 = s[i1];
  float m = fmaxf(v0, v1);
  float e0 = __expf(v0 - m), e1 = __expf(v1 - m);
  float inv = 1.f / (e0 + e1);
  mult[(size_t)idx * 2 + 0] = e0 * inv;
  mult[(size_t)idx * 2 + 1] = e1 * inv;
  sel[(size_t)idx * 2 + 0] = (float)i0;
  sel[(size_t)idx * 2 + 1] = (float)i1;
}

// ---------------------------------------------------------------------------
extern "C" void kernel_launch(void* const* d_in, const int* in_sizes, int n_in,
                              void* d_out, int out_size, void* d_ws, size_t ws_size,
                              hipStream_t stream) {
  (void)in_sizes; (void)n_in; (void)out_size; (void)ws_size;

  const float* x     = (const float*)d_in[0];
  const float* hn    = (const float*)d_in[1];   // [1,B,GH]
  const float* w_ih  = (const float*)d_in[2];   // [3GH,H]
  const float* w_hh  = (const float*)d_in[3];   // [3GH,GH]
  const float* exprw = (const float*)d_in[4];   // [GH,H]
  float* out = (float*)d_out;

  char* ws = (char*)d_ws;
  float*          XI     = (float*)(ws + WS_XI);
  float*          ROUT   = (float*)(ws + WS_ROUT);
  unsigned short* WHH    = (unsigned short*)(ws + WS_WHH);
  unsigned short* WEXP   = (unsigned short*)(ws + WS_WEXP);
  float*          NORMS  = (float*)(ws + WS_NORM);
  float*          PENACC = (float*)(ws + WS_PEN);

  k_init_convert<<<1024, 256, 0, stream>>>(w_hh, exprw, WHH, WEXP, PENACC,
                                           out + OFF_ELOSS);
  k_gemm_bf16<<<dim3(BS / 64, N3 / 128), 256, 0, stream>>>(x, w_ih, XI, N3, Hh);
  k_gemm_bf16<<<dim3(BS / 64, GH / 128), 256, 0, stream>>>(x, exprw,
                                                           out + OFF_EXPR, GH, Hh);
  k_gru_scan<<<1, 1024, 0, stream>>>(XI, WHH, hn, ROUT, out + OFF_HLAST);
  k_row_norms<<<GH * 32 / 256, 256, 0, stream>>>(exprw, NORMS);
  k_expr_loss<<<512, 256, 0, stream>>>(WEXP, NORMS, out + OFF_ELOSS);
  k_gram_cos<<<BS / 8, 256, 0, stream>>>(ROUT, out + OFF_EXPR, out + OFF_COS, PENACC);
  k_finalize_pen<<<1, 1, 0, stream>>>(PENACC, out + OFF_PEN);
  k_topk_softmax<<<BS / 256, 256, 0, stream>>>(out + OFF_COS, out + OFF_PEN,
                                               out + OFF_MULT, out + OFF_SEL);
}
